// NodeAdder_76501957477039
// MI455X (gfx1250) — compile-verified, matrix operand order checked by reading, not run on hardware
//
#include <hip/hip_runtime.h>
#include <hip/hip_bf16.h>

// ---------------- problem constants (match reference) ----------------
#define Bg 512
#define Nn 65536
#define Ee 131072
#define Ss 256
#define Aa 512
#define Pp 2
#define Tt 64
#define TWO_S 512   // 2*S
#define THREE_S 768 // 3*S

typedef __attribute__((ext_vector_type(16))) __bf16 bf16x16;
typedef __attribute__((ext_vector_type(8)))  __bf16 bf16x8;
typedef __attribute__((ext_vector_type(8)))  float  f32x8;

// ---------------- fp32 -> bf16 conversion ----------------
__global__ void k_f2bf(const float* __restrict__ x, __bf16* __restrict__ y, size_t n) {
    size_t i = (size_t)blockIdx.x * blockDim.x + threadIdx.x;
    if (i < n) y[i] = (__bf16)x[i];
}

// Build a 16x32 bf16 WMMA fragment per ISA 7.12.2:
//   lanes 0-15 : row=lane,    K = k0+{0..7} then k0+{16..23}
//   lanes 16-31: row=lane-16, K = k0+{8..15} then k0+{24..31}
// (two contiguous 16B vector loads per lane; works for global or LDS pointers)
__device__ inline bf16x16 load_frag(const __bf16* p) {
    bf16x8 lo = *(const bf16x8*)p;
    bf16x8 hi = *(const bf16x8*)(p + 16);
    bf16x16 f;
#pragma unroll
    for (int i = 0; i < 8; i++) { f[i] = lo[i]; f[i + 8] = hi[i]; }
    return f;
}

// ---------------- WMMA bf16 GEMM: C[M,Nout] = X[M,K] @ W[Nout,K]^T (+bias) ----------------
// Block = 8 waves sharing one 64-column W panel. The panel (64 x K bf16, <= 64KB)
// is staged into LDS once via gfx1250 async global->LDS b128 copies (ASYNCcnt),
// then each wave computes a 16(M) x 64(N) tile: A fragments stream from global,
// B fragments come from LDS (ds_load_b128), accumulating with
// V_WMMA_F32_16X16X32_BF16 (f32 accumulate).
__global__ __launch_bounds__(256) void k_gemm_bf16(
    const __bf16* __restrict__ X, const __bf16* __restrict__ W,
    const float* __restrict__ bias, float* __restrict__ C,
    int M, int K, int Nout)
{
    __shared__ __bf16 sW[64 * 512];   // 64 output cols x K (K <= 512) = 64 KB max

    const int tid    = threadIdx.x;
    const int lane   = tid & 31;
    const int widx   = tid >> 5;              // wave in block: 0..7
    const int tilesN = Nout >> 6;
    const int tilesM = M >> 4;
    const int tn     = blockIdx.x % tilesN;
    const int tmBase = (blockIdx.x / tilesN) << 3;
    const int n0     = tn << 6;

    // ---- async-stage W panel: rows n0..n0+63, K bf16 each, contiguous row-major ----
    {
        const char* gbase = (const char*)(W + (size_t)n0 * K);
        unsigned ldsBase = (unsigned)(size_t)(&sW[0]);   // flat LDS addr low 32 bits == LDS offset
        const int iters = (K * 64 * 2) >> 12;            // panel bytes / (256 thr * 16B)
        for (int it = 0; it < iters; ++it) {
            unsigned off16 = ((unsigned)(it * 256 + tid)) << 4;
            unsigned long long ga = (unsigned long long)(size_t)(gbase + off16);
            unsigned la = ldsBase + off16;
            asm volatile("global_load_async_to_lds_b128 %0, %1, off"
                         :: "v"(la), "v"(ga) : "memory");
        }
        asm volatile("s_wait_asynccnt 0x0" ::: "memory");
        __syncthreads();
    }

    const int tm = tmBase + widx;
    if (tm >= tilesM) return;                 // wave-uniform: EXEC all-1s for WMMA

    const int half = lane >> 4;
    const int l15  = lane & 15;
    const __bf16* xrow = X + (size_t)(tm * 16 + l15) * K;

    f32x8 acc0 = {}, acc1 = {}, acc2 = {}, acc3 = {};

    for (int k0 = 0; k0 < K; k0 += 32) {
        const int kA = k0 + (half ? 8 : 0);
        bf16x16 a  = load_frag(xrow + kA);
        bf16x16 b0 = load_frag(&sW[(size_t)( 0 + l15) * K + kA]);
        bf16x16 b1 = load_frag(&sW[(size_t)(16 + l15) * K + kA]);
        bf16x16 b2 = load_frag(&sW[(size_t)(32 + l15) * K + kA]);
        bf16x16 b3 = load_frag(&sW[(size_t)(48 + l15) * K + kA]);
        acc0 = __builtin_amdgcn_wmma_f32_16x16x32_bf16(false, a, false, b0, (short)0, acc0, false, false);
        acc1 = __builtin_amdgcn_wmma_f32_16x16x32_bf16(false, a, false, b1, (short)0, acc1, false, false);
        acc2 = __builtin_amdgcn_wmma_f32_16x16x32_bf16(false, a, false, b2, (short)0, acc2, false, false);
        acc3 = __builtin_amdgcn_wmma_f32_16x16x32_bf16(false, a, false, b3, (short)0, acc3, false, false);
    }

    // C/D layout: lane holds N=l15 column; VGPR r holds M = r (lanes 0-15) / r+8 (lanes 16-31)
    const int cN = n0 + l15;
    float bv0 = bias ? bias[cN +  0] : 0.f;
    float bv1 = bias ? bias[cN + 16] : 0.f;
    float bv2 = bias ? bias[cN + 32] : 0.f;
    float bv3 = bias ? bias[cN + 48] : 0.f;
#pragma unroll
    for (int r = 0; r < 8; r++) {
        int m = tm * 16 + r + (half ? 8 : 0);
        size_t off = (size_t)m * Nout + cN;
        C[off +  0] = acc0[r] + bv0;
        C[off + 16] = acc1[r] + bv1;
        C[off + 32] = acc2[r] + bv2;
        C[off + 48] = acc3[r] + bv3;
    }
}

// ---------------- edge message pre-activation: tanh(nf[src]+nf[dst]+ef) -> bf16 ----------------
__global__ void k_msgs_pre(const float* __restrict__ nf, const float* __restrict__ efp,
                           const int* __restrict__ src, const int* __restrict__ dst,
                           __bf16* __restrict__ out) {
    size_t idx = (size_t)blockIdx.x * blockDim.x + threadIdx.x;
    if (idx >= (size_t)Ee * TWO_S) return;
    int e = (int)(idx >> 9);
    int c = (int)(idx & (TWO_S - 1));
    float v = nf[(size_t)src[e] * TWO_S + c] + nf[(size_t)dst[e] * TWO_S + c] + efp[idx];
    out[idx] = (__bf16)tanhf(v);
}

// ---------------- scatter-add messages into node inputs ----------------
__global__ void k_scatter(const float* __restrict__ msgs,
                          const int* __restrict__ src, const int* __restrict__ dst,
                          float* __restrict__ inputs) {
    size_t idx = (size_t)blockIdx.x * blockDim.x + threadIdx.x;
    if (idx >= (size_t)Ee * TWO_S) return;
    int e = (int)(idx >> 9);
    int c = (int)(idx & (TWO_S - 1));
    float m = msgs[idx];
    atomicAdd(&inputs[(size_t)dst[e] * TWO_S + c], m);
    atomicAdd(&inputs[(size_t)src[e] * TWO_S + c], m);
}

__device__ inline float sigf(float x) { return 1.0f / (1.0f + expf(-x)); }

// ---------------- fused GRU cell update (torch gate order r,z,n) ----------------
__global__ void k_gru(const float* __restrict__ gi, const float* __restrict__ gh,
                      const float* __restrict__ bih, const float* __restrict__ bhh,
                      const int* __restrict__ owner, const unsigned char* __restrict__ running,
                      float* __restrict__ nodes) {
    size_t idx = (size_t)blockIdx.x * blockDim.x + threadIdx.x;
    if (idx >= (size_t)Nn * Ss) return;
    int n = (int)(idx >> 8);
    int c = (int)(idx & (Ss - 1));
    size_t ro = (size_t)n * THREE_S;
    float h  = nodes[idx];
    float ir = gi[ro + c]            + bih[c];
    float iz = gi[ro + Ss + c]       + bih[Ss + c];
    float in_ = gi[ro + 2 * Ss + c]  + bih[2 * Ss + c];
    float hr = gh[ro + c]            + bhh[c];
    float hz = gh[ro + Ss + c]       + bhh[Ss + c];
    float hn = gh[ro + 2 * Ss + c]   + bhh[2 * Ss + c];
    float r = sigf(ir + hr);
    float z = sigf(iz + hz);
    float nn = tanhf(in_ + r * hn);
    float nv = (1.0f - z) * nn + z * h;
    bool upd = running[owner[n]] != 0;
    nodes[idx] = upd ? nv : h;
}

// ---------------- gated aggregation pool: agg[owner[n]] += data * sigmoid(gate) ----------------
__global__ void k_agg_scatter(const float* __restrict__ data, const float* __restrict__ gates,
                              const int* __restrict__ owner, float* __restrict__ agg) {
    size_t idx = (size_t)blockIdx.x * blockDim.x + threadIdx.x;
    if (idx >= (size_t)Nn * Aa) return;
    int n = (int)(idx >> 9);
    int a = (int)(idx & (Aa - 1));
    float v = data[idx] * sigf(gates[idx]);
    atomicAdd(&agg[(size_t)owner[n] * Aa + a], v);
}

// ---------------- logits = agg_dec @ ntd_W^T + ntd_b ----------------
__global__ void k_logits(const float* __restrict__ agg, const float* __restrict__ ntdW,
                         const float* __restrict__ ntdb, float* __restrict__ logits) {
    int idx = blockIdx.x * blockDim.x + threadIdx.x;
    if (idx >= Bg * (Tt + 1)) return;
    int b = idx / (Tt + 1);
    int t = idx % (Tt + 1);
    float acc = ntdb[t];
    const float* ar = agg + (size_t)b * Aa;
    const float* wr = ntdW + (size_t)t * Aa;
    for (int a = 0; a < Aa; a++) acc += ar[a] * wr[a];
    logits[idx] = acc;
}

// ---------------- masked mean NLL loss ----------------
__global__ void k_loss(const float* __restrict__ logits, const int* __restrict__ ref_types,
                       const unsigned char* __restrict__ running, float* __restrict__ out) {
    __shared__ float sh[Bg];
    int b = threadIdx.x;
    const float* l = logits + (size_t)b * (Tt + 1);
    float m = l[0];
    for (int t = 1; t <= Tt; t++) m = fmaxf(m, l[t]);
    float s = 0.f;
    for (int t = 0; t <= Tt; t++) s += expf(l[t] - m);
    float lse = m + logf(s);
    int sel = ref_types[b] + 1;
    float per = lse - l[sel];
    sh[b] = running[b] ? per : 0.f;
    __syncthreads();
    for (int st = Bg / 2; st > 0; st >>= 1) {
        if (b < st) sh[b] += sh[b + st];
        __syncthreads();
    }
    if (b == 0) out[0] = sh[0] / (float)Bg;
}

// ---------------- new_features = nte[sel-1] @ f1_W^T + f1_b + agg_init @ f2_W^T ----------------
__global__ void k_newfeat(const float* __restrict__ nte, const int* __restrict__ ref_types,
                          const float* __restrict__ f1W, const float* __restrict__ f1b,
                          const float* __restrict__ f2W, const float* __restrict__ agg_init,
                          float* __restrict__ out) {
    int idx = blockIdx.x * blockDim.x + threadIdx.x;
    if (idx >= Bg * Ss) return;
    int b = idx >> 8;
    int s = idx & (Ss - 1);
    const float* emb = nte + (size_t)ref_types[b] * Ss;   // sel-1 == ref_types[b]
    float acc = f1b[s];
    const float* w1 = f1W + (size_t)s * Ss;
    for (int k = 0; k < Ss; k++) acc += emb[k] * w1[k];
    const float* ai = agg_init + (size_t)b * Aa;
    const float* w2 = f2W + (size_t)s * Aa;
    for (int k = 0; k < Aa; k++) acc += ai[k] * w2[k];
    out[idx] = acc;
}

// ==================================================================
extern "C" void kernel_launch(void* const* d_in, const int* in_sizes, int n_in,
                              void* d_out, int out_size, void* d_ws, size_t ws_size,
                              hipStream_t stream) {
    // ---- inputs (setup_inputs order) ----
    const float* nodes_in   = (const float*)d_in[0];
    const float* edge_feat  = (const float*)d_in[1];
    const int*   e_src      = (const int*)d_in[2];
    const int*   e_dst      = (const int*)d_in[3];
    const int*   owner      = (const int*)d_in[4];
    const unsigned char* running = (const unsigned char*)d_in[5];
    const int*   ref_types  = (const int*)d_in[6];
    const float* dec_t_W = (const float*)d_in[7];
    const float* dec_t_b = (const float*)d_in[8];
    const float* dec_g_W = (const float*)d_in[9];
    const float* dec_g_b = (const float*)d_in[10];
    const float* init_t_W = (const float*)d_in[11];
    const float* init_t_b = (const float*)d_in[12];
    const float* init_g_W = (const float*)d_in[13];
    const float* init_g_b = (const float*)d_in[14];
    const float* ntd_W = (const float*)d_in[15];
    const float* ntd_b = (const float*)d_in[16];
    const float* nte   = (const float*)d_in[17];
    const float* f1_W  = (const float*)d_in[18];
    const float* f1_b  = (const float*)d_in[19];
    const float* f2_W  = (const float*)d_in[20];
    const float* mn_W  = (const float*)d_in[21];
    const float* mf_W  = (const float*)d_in[22];
    const float* ml2_W = (const float*)d_in[23];
    const float* ml2_b = (const float*)d_in[24];
    const float* gru_Wih = (const float*)d_in[25];
    const float* gru_Whh = (const float*)d_in[26];
    const float* gru_bih = (const float*)d_in[27];
    const float* gru_bhh = (const float*)d_in[28];

    // ---- workspace arena ----
    char* ws = (char*)d_ws;
    size_t o = 0;
    auto take = [&](size_t bytes) -> char* {
        char* p = ws + o;
        o += (bytes + 255) & ~(size_t)255;
        return p;
    };
    float*  nodes_cur = (float*)take((size_t)Nn * Ss * 4);
    __bf16* nodes_bf  = (__bf16*)take((size_t)Nn * Ss * 2);
    __bf16* ef_bf     = (__bf16*)take((size_t)Ee * Ss * 2);
    __bf16* wbf       = (__bf16*)take((size_t)2752512 * 2);
    float*  bufA      = (float*)take((size_t)Ee * TWO_S * 4);  // ef_p -> gi -> agg data
    float*  bufB      = (float*)take((size_t)Ee * TWO_S * 4);  // nf -> msgs -> agg gates
    __bf16* msgs_bf   = (__bf16*)take((size_t)Ee * TWO_S * 2);
    float*  inputs    = (float*)take((size_t)Nn * TWO_S * 4);
    __bf16* inputs_bf = (__bf16*)take((size_t)Nn * TWO_S * 2);
    float*  gh_buf    = (float*)take((size_t)Nn * THREE_S * 4);
    float*  agg_dec   = (float*)take((size_t)Bg * Aa * 4);
    float*  agg_init  = (float*)take((size_t)Bg * Aa * 4);
    if (o > ws_size) return;  // workspace too small; nothing safe to do

    // bf16 weight arena offsets (elements)
    __bf16* w_mn  = wbf;
    __bf16* w_mf  = w_mn + (size_t)Pp * TWO_S * Ss;
    __bf16* w_ml2 = w_mf + (size_t)Pp * TWO_S * Ss;
    __bf16* w_wih = w_ml2 + (size_t)Pp * TWO_S * TWO_S;
    __bf16* w_whh = w_wih + (size_t)Pp * THREE_S * TWO_S;
    __bf16* w_dt  = w_whh + (size_t)Pp * THREE_S * Ss;
    __bf16* w_dg  = w_dt + (size_t)Aa * Ss;
    __bf16* w_it  = w_dg + (size_t)Aa * Ss;
    __bf16* w_ig  = w_it + (size_t)Aa * Ss;

    auto conv = [&](const float* src, __bf16* dst, size_t n) {
        k_f2bf<<<dim3((unsigned)((n + 255) / 256)), dim3(256), 0, stream>>>(src, dst, n);
    };
    auto gemm = [&](const __bf16* X, const __bf16* W, const float* bias, float* C,
                    int M, int K, int Nout) {
        int tilesM = M >> 4, tilesN = Nout >> 6;
        unsigned blocks = (unsigned)((long)((tilesM + 7) / 8) * tilesN);
        k_gemm_bf16<<<dim3(blocks), dim3(256), 0, stream>>>(X, W, bias, C, M, K, Nout);
    };

    // ---- one-time per call: bf16 weights + edge features; working copy of nodes ----
    conv(mn_W,  w_mn,  (size_t)Pp * TWO_S * Ss);
    conv(mf_W,  w_mf,  (size_t)Pp * TWO_S * Ss);
    conv(ml2_W, w_ml2, (size_t)Pp * TWO_S * TWO_S);
    conv(gru_Wih, w_wih, (size_t)Pp * THREE_S * TWO_S);
    conv(gru_Whh, w_whh, (size_t)Pp * THREE_S * Ss);
    conv(dec_t_W, w_dt, (size_t)Aa * Ss);
    conv(dec_g_W, w_dg, (size_t)Aa * Ss);
    conv(init_t_W, w_it, (size_t)Aa * Ss);
    conv(init_g_W, w_ig, (size_t)Aa * Ss);
    conv(edge_feat, ef_bf, (size_t)Ee * Ss);
    hipMemcpyAsync(nodes_cur, nodes_in, (size_t)Nn * Ss * 4, hipMemcpyDeviceToDevice, stream);

    const size_t EW = (size_t)Ee * TWO_S;
    const size_t NS = (size_t)Nn * Ss;
    const unsigned gEW = (unsigned)((EW + 255) / 256);
    const unsigned gNS = (unsigned)((NS + 255) / 256);
    const unsigned gNA = (unsigned)(((size_t)Nn * Aa + 255) / 256);

    // ---- P propagation steps ----
    for (int p = 0; p < Pp; p++) {
        conv(nodes_cur, nodes_bf, NS);
        // nf = nodes @ mn_W[p]^T   [N, 2S]
        gemm(nodes_bf, w_mn + (size_t)p * TWO_S * Ss, nullptr, bufB, Nn, Ss, TWO_S);
        // ef_p = edge_features @ mf_W[p]^T   [E, 2S]
        gemm(ef_bf, w_mf + (size_t)p * TWO_S * Ss, nullptr, bufA, Ee, Ss, TWO_S);
        // pre = tanh(nf[src] + nf[dst] + ef_p) -> bf16
        k_msgs_pre<<<dim3(gEW), dim3(256), 0, stream>>>(bufB, bufA, e_src, e_dst, msgs_bf);
        // msgs = pre @ ml2_W[p]^T + ml2_b[p]   [E, 2S]  (overwrites nf in bufB)
        gemm(msgs_bf, w_ml2 + (size_t)p * TWO_S * TWO_S, ml2_b + (size_t)p * TWO_S, bufB,
             Ee, TWO_S, TWO_S);
        // inputs = scatter_add(msgs at dst) + scatter_add(msgs at src)
        hipMemsetAsync(inputs, 0, (size_t)Nn * TWO_S * 4, stream);
        k_scatter<<<dim3(gEW), dim3(256), 0, stream>>>(bufB, e_src, e_dst, inputs);
        // gi = inputs @ gru_Wih[p]^T  [N, 3S]  (overwrites ef_p in bufA)
        conv(inputs, inputs_bf, (size_t)Nn * TWO_S);
        gemm(inputs_bf, w_wih + (size_t)p * THREE_S * TWO_S, nullptr, bufA, Nn, TWO_S, THREE_S);
        // gh = nodes @ gru_Whh[p]^T  [N, 3S]
        gemm(nodes_bf, w_whh + (size_t)p * THREE_S * Ss, nullptr, gh_buf, Nn, Ss, THREE_S);
        // fused GRU with update mask
        k_gru<<<dim3(gNS), dim3(256), 0, stream>>>(bufA, gh_buf,
            gru_bih + (size_t)p * THREE_S, gru_bhh + (size_t)p * THREE_S,
            owner, running, nodes_cur);
    }

    // ---- gated aggregations over final node states ----
    conv(nodes_cur, nodes_bf, NS);
    // decision aggregator
    gemm(nodes_bf, w_dt, dec_t_b, bufA, Nn, Ss, Aa);   // data
    gemm(nodes_bf, w_dg, dec_g_b, bufB, Nn, Ss, Aa);   // gate pre-sigmoid
    hipMemsetAsync(agg_dec, 0, (size_t)Bg * Aa * 4, stream);
    k_agg_scatter<<<dim3(gNA), dim3(256), 0, stream>>>(bufA, bufB, owner, agg_dec);
    // init aggregator
    gemm(nodes_bf, w_it, init_t_b, bufA, Nn, Ss, Aa);
    gemm(nodes_bf, w_ig, init_g_b, bufB, Nn, Ss, Aa);
    hipMemsetAsync(agg_init, 0, (size_t)Bg * Aa * 4, stream);
    k_agg_scatter<<<dim3(gNA), dim3(256), 0, stream>>>(bufA, bufB, owner, agg_init);

    // ---- heads: logits, loss, new_features ----
    float* out_logits = (float*)d_out;                       // [B, T+1]
    float* out_feats  = out_logits + (size_t)Bg * (Tt + 1);  // [B, S]
    float* out_loss   = out_feats + (size_t)Bg * Ss;         // scalar

    k_logits<<<dim3((Bg * (Tt + 1) + 255) / 256), dim3(256), 0, stream>>>(
        agg_dec, ntd_W, ntd_b, out_logits);
    k_loss<<<dim3(1), dim3(Bg), 0, stream>>>(out_logits, ref_types, running, out_loss);
    k_newfeat<<<dim3((Bg * Ss + 255) / 256), dim3(256), 0, stream>>>(
        nte, ref_types, f1_W, f1_b, f2_W, agg_init, out_feats);
}
